// DynamicRNN_28552942584266
// MI455X (gfx1250) — matrix operand order repaired and checked by reading, not compiled, not run on hardware
//
#include <hip/hip_runtime.h>
#include <stdint.h>

// Problem sizes (fixed by the reference)
#define Bn   128
#define Tn   512
#define Dn   256
#define Hn   512
#define G4H  2048          // 4*H
#define Kn   768           // D + H (fused GEMM K)
#define KT   24            // Kn / 32 (bf16 WMMA K=32)
#define NBLK 32            // persistent workgroups (8 row tiles x 4 col groups)
#define LDS_STRIDE 776     // 768 + 8 bf16 pad -> conflict-free ds_load_b128

typedef __bf16 bf16_t;
typedef __attribute__((ext_vector_type(8)))  __bf16 v8bf;
typedef __attribute__((ext_vector_type(16))) __bf16 v16bf;
typedef __attribute__((ext_vector_type(8)))  float  v8f;

__device__ __forceinline__ float sigmoidf_(float x) {
  return 1.0f / (1.0f + __expf(-x));
}

// ---------------------------------------------------------------------------
// Kernel 1: transpose + convert weights to bf16, column-major over K.
// Wt[n*768 + k] = (k < 256 ? Wx[k, n] : Wh[k-256, n]) as bf16.
// Tiled 32x32 transpose through LDS for coalesced read AND write.
// grid = (24 k-tiles, 64 n-tiles), block = (32, 8)
// ---------------------------------------------------------------------------
__global__ void prep_weights_kernel(const float* __restrict__ Wx,
                                    const float* __restrict__ Wh,
                                    bf16_t* __restrict__ Wt) {
  __shared__ float tile[32][33];
  const int k0 = blockIdx.x * 32;
  const int n0 = blockIdx.y * 32;
  const int tx = threadIdx.x;       // 0..31  (n for read, k for write)
  const int ty = threadIdx.y;       // 0..7

#pragma unroll
  for (int kk = 0; kk < 4; ++kk) {
    const int kl = ty * 4 + kk;     // 0..31
    const int k  = k0 + kl;
    const int n  = n0 + tx;
    float v = (k < Dn) ? Wx[(size_t)k * G4H + n]
                       : Wh[(size_t)(k - Dn) * G4H + n];
    tile[kl][tx] = v;
  }
  __syncthreads();
#pragma unroll
  for (int kk = 0; kk < 4; ++kk) {
    const int r = ty * 4 + kk;      // local n
    const int n = n0 + r;
    Wt[(size_t)n * Kn + k0 + tx] = (bf16_t)tile[tx][r];
  }
}

// ---------------------------------------------------------------------------
// Kernel 2: zero h/c state and the grid-barrier counter (h0 = c0 = 0).
// ---------------------------------------------------------------------------
__global__ void init_state_kernel(bf16_t* __restrict__ hstate,
                                  float* __restrict__ cstate,
                                  unsigned int* __restrict__ ctr) {
  const int idx = blockIdx.x * blockDim.x + threadIdx.x;
  if (idx < Bn * Hn) {
    hstate[idx] = (bf16_t)0.0f;
    cstate[idx] = 0.0f;
  }
  if (idx == 0) *ctr = 0u;
}

// ---------------------------------------------------------------------------
// Kernel 3: persistent fused LSTM scan.
//   Per step: z = [x_t | h] @ [Wx; Wh] + b  (bf16 WMMA, f32 accum), then gates.
//   32 WGs x 256 thr (8 waves). WG = (rowTile 0..7, colGroup 0..3).
//   Wave w owns colTile = colGroup*8 + w; accumulates i/f/g/o 16x16 tiles.
//   Grid barrier between steps via monotonic agent-scope atomic counter.
//   NOTE: K loop must NOT be fully unrolled -- constant addresses become
//   t-invariant, LICM hoists all 96 B-fragments and spills them to scratch.
//   unroll_count(4) keeps weight loads as L2-hitting global_load_b128.
// ---------------------------------------------------------------------------
__global__ __launch_bounds__(256, 1)
void lstm_scan_kernel(const float* __restrict__ x,     // [B,T,D] f32
                      const bf16_t* __restrict__ Wt,   // [4H,768] bf16 (K contiguous)
                      const float* __restrict__ bias,  // [4H]
                      float* __restrict__ out,         // outputs|hT|cT
                      bf16_t* __restrict__ hstate,     // [B,H] bf16
                      float* __restrict__ cstate,      // [B,H] f32
                      unsigned int* __restrict__ barrier_ctr) {
  __shared__ bf16_t Atile[16 * LDS_STRIDE];   // [x_t | h_prev] row tile, bf16

  const int tid      = threadIdx.x;
  const int lane     = tid & 31;
  const int wave     = tid >> 5;                 // 0..7
  const int wg       = blockIdx.x;               // 0..31
  const int rowTile  = wg & 7;                   // 8 tiles of 16 batch rows
  const int colGroup = wg >> 3;                  // 0..3
  const int colTile  = colGroup * 8 + wave;      // 0..31 (16 h-cols each)
  const int rowBase  = rowTile * 16;
  const int laneHi   = lane >> 4;                // 0|1
  const int laneLo   = lane & 15;
  const int nCol     = colTile * 16 + laneLo;    // h index 0..511

  // Per-gate weight pointers (B fragment: 16 contiguous K values per lane)
  // and biases; fixed for the whole scan.
  const bf16_t* wptr0 = Wt + (size_t)(0 * Hn + nCol) * Kn + laneHi * 16;
  const bf16_t* wptr1 = Wt + (size_t)(1 * Hn + nCol) * Kn + laneHi * 16;
  const bf16_t* wptr2 = Wt + (size_t)(2 * Hn + nCol) * Kn + laneHi * 16;
  const bf16_t* wptr3 = Wt + (size_t)(3 * Hn + nCol) * Kn + laneHi * 16;
  const float bias0 = bias[0 * Hn + nCol];
  const float bias1 = bias[1 * Hn + nCol];
  const float bias2 = bias[2 * Hn + nCol];
  const float bias3 = bias[3 * Hn + nCol];

  // C/D tile layout: VGPR v -> M = 8*laneHi + v, N = laneLo.
  const int    stBase = (rowBase + 8 * laneHi) * Hn + nCol;  // h/c state index
  const size_t OUT_H  = (size_t)Bn * Tn * Hn;                // hT offset
  const size_t OUT_C  = OUT_H + (size_t)Bn * Hn;             // cT offset

  for (int t = 0; t < Tn; ++t) {
    // ---- stage A = [x_t (f32->bf16) | h_prev (bf16)] into LDS ----
    for (int i = tid; i < 16 * 128; i += 256) {          // x: 16 rows x 256, float2
      const int r  = i >> 7;
      const int d2 = (i & 127) * 2;
      const float2 v = *(const float2*)(x + ((size_t)(rowBase + r) * Tn + t) * Dn + d2);
      Atile[r * LDS_STRIDE + d2]     = (bf16_t)v.x;
      Atile[r * LDS_STRIDE + d2 + 1] = (bf16_t)v.y;
    }
    for (int i = tid; i < 16 * 256; i += 256) {          // h: 16 rows x 512 bf16, dword copy
      const int r  = i >> 8;
      const int j2 = (i & 255) * 2;
      *(uint32_t*)(&Atile[r * LDS_STRIDE + Dn + j2]) =
          *(const uint32_t*)(hstate + (rowBase + r) * Hn + j2);
    }
    // prefetch next step's x tile into caches (global_prefetch_b8)
    if (t + 1 < Tn) {
      const int r   = tid >> 4;
      const int seg = (tid & 15) * 16;
      __builtin_prefetch(x + ((size_t)(rowBase + r) * Tn + (t + 1)) * Dn + seg, 0, 0);
    }
    __syncthreads();

    // ---- K loop: 24 x (A 16x32 bf16) x (B 32x16 bf16) per gate ----
    v8f acc0 = {}, acc1 = {}, acc2 = {}, acc3 = {};
    const bf16_t* arow = Atile + laneLo * LDS_STRIDE + laneHi * 16;
#pragma clang loop unroll_count(4)
    for (int kt = 0; kt < KT; ++kt) {
      const int kg = kt * 32;
      // A fragment: lane row = laneLo; a[0..7]=A[r,kg+16*laneHi+0..7],
      //             a[8..15]=A[r,kg+16*laneHi+8..15]
      v8bf alo = *(const v8bf*)(arow + kg);
      v8bf ahi = *(const v8bf*)(arow + kg + 8);
      v16bf a = __builtin_shufflevector(alo, ahi,
                  0, 1, 2, 3, 4, 5, 6, 7, 8, 9, 10, 11, 12, 13, 14, 15);
      // B fragments: 16 contiguous K values for column nCol (+gate offset)
      v16bf b0 = *(const v16bf*)(wptr0 + kg);
      v16bf b1 = *(const v16bf*)(wptr1 + kg);
      v16bf b2 = *(const v16bf*)(wptr2 + kg);
      v16bf b3 = *(const v16bf*)(wptr3 + kg);
      acc0 = __builtin_amdgcn_wmma_f32_16x16x32_bf16(false, a, false, b0,
                                                     (short)0, acc0, false, false);
      acc1 = __builtin_amdgcn_wmma_f32_16x16x32_bf16(false, a, false, b1,
                                                     (short)0, acc1, false, false);
      acc2 = __builtin_amdgcn_wmma_f32_16x16x32_bf16(false, a, false, b2,
                                                     (short)0, acc2, false, false);
      acc3 = __builtin_amdgcn_wmma_f32_16x16x32_bf16(false, a, false, b3,
                                                     (short)0, acc3, false, false);
    }

    // ---- gate epilogue: each lane owns 8 (batch-row, h-col) cells ----
#pragma unroll
    for (int v = 0; v < 8; ++v) {
      const int   brow = rowBase + 8 * laneHi + v;
      const int   sIdx = stBase + v * Hn;
      const float zi = acc0[v] + bias0;
      const float zf = acc1[v] + bias1;
      const float zg = acc2[v] + bias2;
      const float zo = acc3[v] + bias3;
      const float cprev = cstate[sIdx];
      const float ig = sigmoidf_(zi);
      const float fg = sigmoidf_(zf);
      const float gg = tanhf(zg);
      const float og = sigmoidf_(zo);
      const float cn = fg * cprev + ig * gg;
      const float hn = og * tanhf(cn);
      cstate[sIdx] = cn;
      hstate[sIdx] = (bf16_t)hn;
      out[(size_t)brow * Tn * Hn + (size_t)t * Hn + nCol] = hn;
      if (t == Tn - 1) {
        out[OUT_H + sIdx] = hn;   // hT
        out[OUT_C + sIdx] = cn;   // cT
      }
    }

    // ---- grid-wide barrier (monotonic counter, agent scope) ----
    __threadfence();
    __syncthreads();
    if (tid == 0) {
      __hip_atomic_fetch_add(barrier_ctr, 1u, __ATOMIC_ACQ_REL,
                             __HIP_MEMORY_SCOPE_AGENT);
      const unsigned target = (unsigned)NBLK * (unsigned)(t + 1);
      while (__hip_atomic_load(barrier_ctr, __ATOMIC_ACQUIRE,
                               __HIP_MEMORY_SCOPE_AGENT) < target) {
        __builtin_amdgcn_s_sleep(1);
      }
    }
    __syncthreads();
  }
}

// ---------------------------------------------------------------------------
// Host launch
// ---------------------------------------------------------------------------
extern "C" void kernel_launch(void* const* d_in, const int* in_sizes, int n_in,
                              void* d_out, int out_size, void* d_ws, size_t ws_size,
                              hipStream_t stream) {
  const float* x    = (const float*)d_in[0];   // [B,T,D]
  const float* Wx   = (const float*)d_in[1];   // [D,4H]
  const float* Wh   = (const float*)d_in[2];   // [H,4H]
  const float* bias = (const float*)d_in[3];   // [4H]
  float* out = (float*)d_out;                  // outputs|hT|cT

  // Workspace layout
  char* ws = (char*)d_ws;
  const size_t wtBytes = (size_t)G4H * Kn * sizeof(bf16_t);   // 3,145,728
  const size_t hBytes  = (size_t)Bn * Hn * sizeof(bf16_t);    //   131,072
  const size_t cBytes  = (size_t)Bn * Hn * sizeof(float);     //   262,144
  bf16_t* Wt      = (bf16_t*)ws;
  bf16_t* hstate  = (bf16_t*)(ws + wtBytes);
  float*  cstate  = (float*)(ws + wtBytes + hBytes);
  unsigned int* ctr = (unsigned int*)(ws + wtBytes + hBytes + cBytes);

  // 1) weights -> bf16, transposed so K is contiguous per output column
  prep_weights_kernel<<<dim3(Kn / 32, G4H / 32), dim3(32, 8), 0, stream>>>(Wx, Wh, Wt);
  // 2) h0 = c0 = 0, barrier counter = 0
  init_state_kernel<<<(Bn * Hn + 255) / 256, 256, 0, stream>>>(hstate, cstate, ctr);
  // 3) persistent fused scan
  lstm_scan_kernel<<<NBLK, 256, 0, stream>>>(x, Wt, bias, out, hstate, cstate, ctr);
}
